// SlowMha_26809185862036
// MI455X (gfx1250) — matrix operand order repaired and checked by reading, not compiled, not run on hardware
//
#include <hip/hip_runtime.h>
#include <hip/hip_bf16.h>
#include <math.h>

// ---------------------------------------------------------------------------
// MHA forward for MI455X (gfx1250): bf16 WMMA + Tensor Data Mover staging.
// ---------------------------------------------------------------------------

typedef __attribute__((ext_vector_type(16))) __bf16 v16bf;
typedef __attribute__((ext_vector_type(8)))  __bf16 v8bf;
typedef __attribute__((ext_vector_type(8)))  float  v8f;
typedef __attribute__((ext_vector_type(4)))  unsigned int v4u;
typedef __attribute__((ext_vector_type(8)))  int v8i;
typedef __attribute__((ext_vector_type(4)))  int v4i;

#define DIMX     128
#define NHEADS   16
#define HEAD_DIM 128
#define SEQ      2048
#define BATCH    2

#if defined(__HIP_DEVICE_COMPILE__) && __has_builtin(__builtin_amdgcn_tensor_load_to_lds)
#define HAVE_TDM 1
#else
#define HAVE_TDM 0
#endif

#if HAVE_TDM
// Issue one TDM 2D tile load (bf16 elements): global[rows x cols] -> LDS,
// descriptor packed per CDNA5 ISA 8.3/8.4 (D# groups 0..3, groups 2/3 unused).
__device__ __forceinline__ void tdm_load_2d(unsigned lds_addr, const void* gaddr,
                                            unsigned rows, unsigned cols,
                                            unsigned row_stride_elems,
                                            unsigned tensor_rows) {
  unsigned long long ga = (unsigned long long)(uintptr_t)gaddr;
  v4u g0;
  g0[0] = 1u;                                   // count=1, no gather/restore
  g0[1] = lds_addr;                             // LDS byte address
  g0[2] = (unsigned)(ga & 0xffffffffu);         // global_addr[31:0]
  g0[3] = (unsigned)((ga >> 32) & 0x1ffffffu)   // global_addr[56:32]
          | (2u << 30);                          // type=2 ("image")
  v8i g1;
  g1[0] = (int)(1u << 16);                      // data_size=1 (2 bytes)
  g1[1] = (int)((cols & 0xffffu) << 16);        // tensor_dim0[15:0] @ bits63:48
  g1[2] = (int)(((cols >> 16) & 0xffffu) |      // tensor_dim0[31:16]
                ((tensor_rows & 0xffffu) << 16)); // tensor_dim1[15:0]
  g1[3] = (int)(((tensor_rows >> 16) & 0xffffu) | // tensor_dim1[31:16]
                ((cols & 0xffffu) << 16));      // tile_dim0
  g1[4] = (int)(rows & 0xffffu);                // tile_dim1 (tile_dim2=0)
  g1[5] = (int)row_stride_elems;                // tensor_dim0_stride[31:0]
  g1[6] = 0;                                    // stride0[47:32] | stride1[15:0]
  g1[7] = 0;                                    // tensor_dim1_stride (unused, 2D)
  v4i z4 = {0, 0, 0, 0};
#if __clang_major__ >= 23
  v8i z8 = {0, 0, 0, 0, 0, 0, 0, 0};
  __builtin_amdgcn_tensor_load_to_lds(g0, g1, z4, z4, z8, 0);
#else
  __builtin_amdgcn_tensor_load_to_lds(g0, g1, z4, z4, 0);
#endif
}
#endif  // HAVE_TDM

// ---- WMMA fragment helpers (layouts per CDNA5 ISA 7.12.2, wave32) ----------

// A-matrix 16x32 bf16: lane L(0-15) row M=L holds K = {0..7, 16..23}+half*8
__device__ __forceinline__ v16bf lds_load_a(const __bf16* lds, int ld, int m0, int k0) {
  int lane = threadIdx.x & 31;
  int hf   = lane >> 4;
  int row  = m0 + (lane & 15);
  const __bf16* p = lds + row * ld + k0 + hf * 8;
  union { v16bf v; v8bf h[2]; } u;
  u.h[0] = *(const v8bf*)(p);       // K = half*8 + 0..7
  u.h[1] = *(const v8bf*)(p + 16);  // K = half*8 + 16..23
  return u.v;
}

// B-matrix 32x16 bf16 loaded from B^T storage [n][k]: lane col N=lane&15,
// lane holds K = {0..7, 16..23}+half*8 (same order as A).
__device__ __forceinline__ v16bf lds_load_bt(const __bf16* lds, int ld, int n0, int k0) {
  int lane = threadIdx.x & 31;
  int hf   = lane >> 4;
  int col  = n0 + (lane & 15);
  const __bf16* p = lds + col * ld + k0 + hf * 8;
  union { v16bf v; v8bf h[2]; } u;
  u.h[0] = *(const v8bf*)(p);
  u.h[1] = *(const v8bf*)(p + 16);
  return u.v;
}

__device__ __forceinline__ v8f wmma_bf16(v16bf a, v16bf b, v8f c) {
  return __builtin_amdgcn_wmma_f32_16x16x32_bf16(false, a, false, b, (short)0, c,
                                                 false, false);
}

// C/D layout: VGPR r holds row (r + 8*(lane>>4)), col = lane&15.

// ---------------------------------------------------------------------------
// Kernel 1: qkv = x @ qkv_proj^T  (+RoPE on q,k), bf16 outputs [b][h][t][d]
// M = B*T = 4096, N = 3*2048 = 6144, K = 128.  64x64 tiles, 4 waves.
// ---------------------------------------------------------------------------
__global__ __launch_bounds__(128)
void qkv_rope_kernel(const float* __restrict__ x, const float* __restrict__ proj,
                     __bf16* __restrict__ qB, __bf16* __restrict__ kB,
                     __bf16* __restrict__ vB) {
  __shared__ __align__(16) __bf16 ldsX[64 * DIMX];
  __shared__ __align__(16) __bf16 ldsW[64 * DIMX];

  const int n0  = blockIdx.x * 64;          // 0..6143
  const int m0  = blockIdx.y * 64;          // 0..4095
  const int tid = threadIdx.x;
  const int wv  = tid >> 5;
  const int lane = tid & 31, hf = lane >> 4, ln = lane & 15;

  // stage x tile fp32->bf16
  {
    const float4* src = (const float4*)(x + (size_t)m0 * DIMX);
    for (int i = tid; i < 64 * DIMX / 4; i += 128) {
      float4 f = src[i];
      union { __bf16 b[4]; uint2 u; } pk;
      pk.b[0] = (__bf16)f.x; pk.b[1] = (__bf16)f.y;
      pk.b[2] = (__bf16)f.z; pk.b[3] = (__bf16)f.w;
      *(uint2*)(ldsX + i * 4) = pk.u;
    }
  }
  // stage weight tile rows n0..n0+63 of proj viewed as [6144][128]
  {
    const float4* src = (const float4*)(proj + (size_t)n0 * DIMX);
    for (int i = tid; i < 64 * DIMX / 4; i += 128) {
      float4 f = src[i];
      union { __bf16 b[4]; uint2 u; } pk;
      pk.b[0] = (__bf16)f.x; pk.b[1] = (__bf16)f.y;
      pk.b[2] = (__bf16)f.z; pk.b[3] = (__bf16)f.w;
      *(uint2*)(ldsW + i * 4) = pk.u;
    }
  }
  __syncthreads();

  const int mrow = wv * 16;
  v8f acc[4] = {};
#pragma unroll
  for (int kc = 0; kc < 4; ++kc) {
    v16bf afr = lds_load_a(ldsX, DIMX, mrow, kc * 32);
#pragma unroll
    for (int nt = 0; nt < 4; ++nt) {
      v16bf bfr = lds_load_bt(ldsW, DIMX, nt * 16, kc * 32);
      acc[nt] = wmma_bf16(afr, bfr, acc[nt]);
    }
  }

  // RoPE + store. Pair (2i,2i+1) lives in adjacent lanes -> shfl_xor(1).
  const float kLog = 9.2103403720f / 128.0f;  // ln(10000)/rot_dim
#pragma unroll
  for (int nt = 0; nt < 4; ++nt) {
    int j    = n0 + nt * 16 + ln;   // 0..6143 (a,h uniform per 16-col tile)
    int aidx = j >> 11;             // 0=q 1=k 2=v
    int hh   = (j >> 7) & 15;
    int d    = j & 127;
    float speed = __expf(-(float)(d & ~1) * kLog);
#pragma unroll
    for (int r = 0; r < 8; ++r) {
      int m  = m0 + mrow + r + hf * 8;
      int t  = m & (SEQ - 1);
      int bi = m >> 11;
      float val     = acc[nt][r];
      float partner = __shfl_xor(val, 1, 32);
      float y;
      if (aidx < 2) {
        float ang = (float)t * speed;
        float c = __cosf(ang), s = __sinf(ang);
        y = (d & 1) ? (partner * s + val * c)    // yi = xr*s + xi*c
                    : (val * c - partner * s);   // yr = xr*c - xi*s
      } else {
        y = val;
      }
      __bf16* dst = (aidx == 0) ? qB : (aidx == 1) ? kB : vB;
      dst[(((size_t)(bi * NHEADS + hh)) * SEQ + t) * HEAD_DIM + d] = (__bf16)y;
    }
  }
}

// ---------------------------------------------------------------------------
// Kernel 2: causal attention per (b,h,64-row block).
// Phase 1: logits = scale*Q@K^T (masked) -> attn buffer, online row max/sum.
//          Q/K tiles staged into LDS by the Tensor Data Mover.
// Phase 2: normalize probs in attn buffer, O += P@V via WMMA.
// ---------------------------------------------------------------------------
__global__ __launch_bounds__(128)
void attn_kernel(const __bf16* __restrict__ qB, const __bf16* __restrict__ kB,
                 const __bf16* __restrict__ vB, float* __restrict__ attn,
                 float* __restrict__ oF) {
  __shared__ __align__(16) __bf16 ldsX[64 * DIMX];  // Q staging, then P staging
  __shared__ __align__(16) __bf16 ldsB[64 * DIMX];  // K tile [s][d] / V^T [d][s]

  const int nmb = SEQ / 64;
  const int blk = blockIdx.x;
  const int mblk = blk % nmb;
  const int h    = (blk / nmb) % NHEADS;
  const int bi   = blk / (nmb * NHEADS);
  const int m0   = mblk * 64;

  const int tid = threadIdx.x;
  const int wv  = tid >> 5;
  const int lane = tid & 31, hf = lane >> 4, ln = lane & 15;
  const int mrow = wv * 16;

  const size_t headOff = (size_t)(bi * NHEADS + h) * SEQ * HEAD_DIM;
  const __bf16* qh = qB + headOff;
  const __bf16* kh = kB + headOff;
  const __bf16* vh = vB + headOff;
  float* attnh = attn + ((size_t)(bi * NHEADS + h) * SEQ + m0) * SEQ;

  // ---- stage Q tile via TDM, keep fragments in registers ----
#if HAVE_TDM
  if (wv == 0) {
    tdm_load_2d((unsigned)(uintptr_t)ldsX, qh + (size_t)m0 * HEAD_DIM,
                64, HEAD_DIM, HEAD_DIM, SEQ);
    __builtin_amdgcn_s_wait_tensorcnt(0);
  }
#else
  {
    const uint4* src = (const uint4*)(qh + (size_t)m0 * HEAD_DIM);
    uint4* dst = (uint4*)ldsX;
    for (int i = tid; i < 64 * DIMX / 8; i += 128) dst[i] = src[i];
  }
#endif
  __syncthreads();
  v16bf qfrag[4];
#pragma unroll
  for (int kc = 0; kc < 4; ++kc) qfrag[kc] = lds_load_a(ldsX, DIMX, mrow, kc * 32);

  float mrs[8], lrs[8];
#pragma unroll
  for (int r = 0; r < 8; ++r) { mrs[r] = -INFINITY; lrs[r] = 0.f; }

  const int diagBlk = mblk;
  const float scale = 0.08838834764831845f;  // HEAD_DIM^-0.5

  // ---------------- Phase 1: logits + online softmax stats -----------------
  for (int sb = 0; sb <= diagBlk; ++sb) {
    const int s0 = sb * 64;
#if HAVE_TDM
    if (wv == 0) {
      tdm_load_2d((unsigned)(uintptr_t)ldsB, kh + (size_t)s0 * HEAD_DIM,
                  64, HEAD_DIM, HEAD_DIM, SEQ);
      __builtin_amdgcn_s_wait_tensorcnt(0);
    }
#else
    {
      const uint4* src = (const uint4*)(kh + (size_t)s0 * HEAD_DIM);
      uint4* dst = (uint4*)ldsB;
      for (int i = tid; i < 64 * DIMX / 8; i += 128) dst[i] = src[i];
    }
#endif
    __syncthreads();

    v8f acc[4] = {};
#pragma unroll
    for (int kc = 0; kc < 4; ++kc) {
#pragma unroll
      for (int nt = 0; nt < 4; ++nt) {
        v16bf bfr = lds_load_bt(ldsB, DIMX, nt * 16, kc * 32);  // B[k=d][n=key]
        acc[nt] = wmma_bf16(qfrag[kc], bfr, acc[nt]);
      }
    }
    // mask, scale, spill raw logits, update stats
#pragma unroll
    for (int nt = 0; nt < 4; ++nt) {
      int scol = s0 + nt * 16 + ln;
#pragma unroll
      for (int r = 0; r < 8; ++r) {
        int row  = mrow + r + hf * 8;
        int trow = m0 + row;
        float v = acc[nt][r] * scale;
        if (scol > trow) v = -INFINITY;
        attnh[(size_t)row * SEQ + scol] = v;
        acc[nt][r] = v;
      }
    }
#pragma unroll
    for (int r = 0; r < 8; ++r) {
      float mx = fmaxf(fmaxf(acc[0][r], acc[1][r]), fmaxf(acc[2][r], acc[3][r]));
#pragma unroll
      for (int off = 1; off < 16; off <<= 1) mx = fmaxf(mx, __shfl_xor(mx, off, 32));
      float mnew = fmaxf(mrs[r], mx);
      float sum = __expf(acc[0][r] - mnew) + __expf(acc[1][r] - mnew) +
                  __expf(acc[2][r] - mnew) + __expf(acc[3][r] - mnew);
#pragma unroll
      for (int off = 1; off < 16; off <<= 1) sum += __shfl_xor(sum, off, 32);
      lrs[r] = lrs[r] * __expf(mrs[r] - mnew) + sum;
      mrs[r] = mnew;
    }
    __syncthreads();
  }

  float linv[8];
#pragma unroll
  for (int r = 0; r < 8; ++r) linv[r] = 1.0f / lrs[r];

  // ---------------- Phase 2: normalize probs + O = P@V ---------------------
  __bf16* pstage = ldsX + wv * (16 * 64);  // per-wave 16x64 P tile
  v8f oacc[8] = {};

  for (int sb = 0; sb <= diagBlk; ++sb) {
    const int s0 = sb * 64;
    // stage V transposed: ldsB[d][s]  (contiguous global reads, strided LDS)
    for (int i = tid; i < 64 * DIMX / 8; i += 128) {
      int s  = i >> 4;
      int d8 = (i & 15) * 8;
      union { uint4 u; __bf16 b[8]; } pk;
      pk.u = *(const uint4*)(vh + (size_t)(s0 + s) * HEAD_DIM + d8);
#pragma unroll
      for (int j = 0; j < 8; ++j) ldsB[(d8 + j) * 64 + s] = pk.b[j];
    }
    // re-read logits, normalize, write probs, stage bf16 P
#pragma unroll
    for (int nt = 0; nt < 4; ++nt) {
      int scol = s0 + nt * 16 + ln;
#pragma unroll
      for (int r = 0; r < 8; ++r) {
        int row = mrow + r + hf * 8;
        float lg = attnh[(size_t)row * SEQ + scol];
        float p  = __expf(lg - mrs[r]) * linv[r];
        attnh[(size_t)row * SEQ + scol] = p;
        pstage[(r + hf * 8) * 64 + nt * 16 + ln] = (__bf16)p;
      }
    }
    __syncthreads();  // V tile visible to all waves
#pragma unroll
    for (int kc = 0; kc < 2; ++kc) {
      v16bf pfr = lds_load_a(pstage, 64, 0, kc * 32);
#pragma unroll
      for (int nt = 0; nt < 8; ++nt) {
        v16bf vfr = lds_load_bt(ldsB, 64, nt * 16, kc * 32);  // B[k=s][n=d]
        oacc[nt] = wmma_bf16(pfr, vfr, oacc[nt]);
      }
    }
    __syncthreads();  // all waves done with ldsB before next fill
  }

  // zero-fill fully masked attn blocks (strict upper triangle)
  for (int sb = diagBlk + 1; sb < nmb; ++sb) {
    const int s0 = sb * 64;
    for (int i = tid; i < 64 * 64 / 4; i += 128) {
      int row = i >> 4;
      int c4  = i & 15;
      *(float4*)(attnh + (size_t)row * SEQ + s0 + c4 * 4) =
          make_float4(0.f, 0.f, 0.f, 0.f);
    }
  }

  // store O strip (fp32) to per-head buffer [b][t][h*HD + d]
#pragma unroll
  for (int nt = 0; nt < 8; ++nt) {
#pragma unroll
    for (int r = 0; r < 8; ++r) {
      int row = m0 + mrow + r + hf * 8;
      int d   = nt * 16 + ln;
      oF[((size_t)(bi * SEQ + row)) * (NHEADS * HEAD_DIM) + h * HEAD_DIM + d] =
          oacc[nt][r];
    }
  }
}

// ---------------------------------------------------------------------------
// Kernel 3: out = oF @ out_w^T.  M=4096, N=128, K=2048.  64-row blocks.
// ---------------------------------------------------------------------------
__global__ __launch_bounds__(128)
void out_proj_kernel(const float* __restrict__ oF, const float* __restrict__ out_w,
                     float* __restrict__ out) {
  __shared__ __align__(16) __bf16 ldsA[64 * 32];
  __shared__ __align__(16) __bf16 ldsW[128 * 32];

  const int m0  = blockIdx.x * 64;
  const int tid = threadIdx.x;
  const int wv  = tid >> 5;
  const int lane = tid & 31, hf = lane >> 4, ln = lane & 15;
  const int mrow = wv * 16;
  const int KD = NHEADS * HEAD_DIM;  // 2048

  v8f acc[8] = {};
  for (int k0 = 0; k0 < KD; k0 += 32) {
    for (int i = tid; i < 64 * 32 / 4; i += 128) {
      int row = i >> 3, c4 = i & 7;
      float4 f = *(const float4*)(oF + (size_t)(m0 + row) * KD + k0 + c4 * 4);
      union { __bf16 b[4]; uint2 u; } pk;
      pk.b[0] = (__bf16)f.x; pk.b[1] = (__bf16)f.y;
      pk.b[2] = (__bf16)f.z; pk.b[3] = (__bf16)f.w;
      *(uint2*)(ldsA + row * 32 + c4 * 4) = pk.u;
    }
    for (int i = tid; i < 128 * 32 / 4; i += 128) {
      int row = i >> 3, c4 = i & 7;
      float4 f = *(const float4*)(out_w + (size_t)row * KD + k0 + c4 * 4);
      union { __bf16 b[4]; uint2 u; } pk;
      pk.b[0] = (__bf16)f.x; pk.b[1] = (__bf16)f.y;
      pk.b[2] = (__bf16)f.z; pk.b[3] = (__bf16)f.w;
      *(uint2*)(ldsW + row * 32 + c4 * 4) = pk.u;
    }
    __syncthreads();
    v16bf afr = lds_load_a(ldsA, 32, mrow, 0);
#pragma unroll
    for (int nt = 0; nt < 8; ++nt) {
      v16bf bfr = lds_load_bt(ldsW, 32, nt * 16, 0);
      acc[nt] = wmma_bf16(afr, bfr, acc[nt]);
    }
    __syncthreads();
  }
#pragma unroll
  for (int nt = 0; nt < 8; ++nt) {
#pragma unroll
    for (int r = 0; r < 8; ++r) {
      int m = m0 + mrow + r + hf * 8;
      out[(size_t)m * DIMX + nt * 16 + ln] = acc[nt][r];
    }
  }
}

// ---------------------------------------------------------------------------
extern "C" void kernel_launch(void* const* d_in, const int* in_sizes, int n_in,
                              void* d_out, int out_size, void* d_ws, size_t ws_size,
                              hipStream_t stream) {
  (void)in_sizes; (void)n_in; (void)out_size; (void)ws_size;
  const float* x     = (const float*)d_in[0];
  const float* proj  = (const float*)d_in[1];
  const float* out_w = (const float*)d_in[2];

  const size_t QKV_ELEMS = (size_t)BATCH * NHEADS * SEQ * HEAD_DIM;  // 8.4M
  __bf16* qB = (__bf16*)d_ws;
  __bf16* kB = qB + QKV_ELEMS;
  __bf16* vB = kB + QKV_ELEMS;
  float*  oF = (float*)(vB + QKV_ELEMS);   // [B][T][NH*HD] fp32 (32 MB)

  float* outp  = (float*)d_out;                            // (B,T,128)
  float* attnp = outp + (size_t)BATCH * SEQ * DIMX;        // (B,NH,T,T)

  qkv_rope_kernel<<<dim3(3 * NHEADS * HEAD_DIM / 64, BATCH * SEQ / 64), 128, 0,
                    stream>>>(x, proj, qB, kB, vB);
  attn_kernel<<<dim3(BATCH * NHEADS * (SEQ / 64)), 128, 0, stream>>>(
      qB, kB, vB, attnp, oF);
  out_proj_kernel<<<dim3(BATCH * SEQ / 64), 128, 0, stream>>>(oF, out_w, outp);
}
